// InvariantPointAttentionV46_66949950210583
// MI455X (gfx1250) — compile-verified, matrix operand order checked by reading, not run on hardware
//
#include <hip/hip_runtime.h>

typedef __bf16 bf16;
typedef __attribute__((ext_vector_type(16))) __bf16 v16bf;
typedef __attribute__((ext_vector_type(8)))  __bf16 v8bf;
typedef __attribute__((ext_vector_type(8)))  float  v8f;
typedef __attribute__((ext_vector_type(4)))  float  v4f;

#define N_SEQ 1024
#define D_S   768
#define N_H   12
#define C_H   64
#define P_D   12      // P*3
#define PD_PAD 32     // padded K for point WMMA
#define K_WP  160     // 144 padded to multiple of 32
#define NCHUNK 4
#define CHK   256
#define LSTR  257     // LDS row stride (bank-conflict padding)

// ---------------------------------------------------------------------------
// WMMA helpers (CDNA5 bf16 16x16x32, f32 accumulate)
// ---------------------------------------------------------------------------
static __device__ inline v8f wmma_b(v16bf a, v16bf b, v8f c) {
  return __builtin_amdgcn_wmma_f32_16x16x32_bf16(false, a, false, b, (short)0, c,
                                                 false, false);
}

// 16x32 bf16 fragment from row-major memory. ISA 16-bit A/B layout:
// lane<16 holds K {0..7,16..23}, lane>=16 holds K {8..15,24..31}; row = lane&15.
static __device__ inline v16bf ld_frag_bf(const bf16* base, int ld, int k0) {
  int lane = threadIdx.x & 31;
  const bf16* p = base + (size_t)(lane & 15) * ld + k0 + ((lane & 16) ? 8 : 0);
  v8bf lo = *(const v8bf*)p;
  v8bf hi = *(const v8bf*)(p + 16);
  v16bf f;
#pragma unroll
  for (int i = 0; i < 8; ++i) { f[i] = lo[i]; f[8 + i] = hi[i]; }
  return f;
}

// Same fragment but converting from f32 source inline.
static __device__ inline v16bf ld_frag_f32(const float* base, int ld, int k0) {
  int lane = threadIdx.x & 31;
  const float* p = base + (size_t)(lane & 15) * ld + k0 + ((lane & 16) ? 8 : 0);
  v4f a0 = *(const v4f*)(p);
  v4f a1 = *(const v4f*)(p + 4);
  v4f b0 = *(const v4f*)(p + 16);
  v4f b1 = *(const v4f*)(p + 20);
  v16bf f;
#pragma unroll
  for (int i = 0; i < 4; ++i) {
    f[i]      = (bf16)a0[i];
    f[4 + i]  = (bf16)a1[i];
    f[8 + i]  = (bf16)b0[i];
    f[12 + i] = (bf16)b1[i];
  }
  return f;
}

// ---------------------------------------------------------------------------
// Small utility kernels
// ---------------------------------------------------------------------------
__global__ void cvt_bf_kernel(const float* __restrict__ s, bf16* __restrict__ d, int n) {
  int i = blockIdx.x * blockDim.x + threadIdx.x;
  if (i < n) d[i] = (bf16)s[i];
}

__global__ void zero_bf_kernel(bf16* __restrict__ p, int n) {
  int i = blockIdx.x * blockDim.x + threadIdx.x;
  if (i < n) p[i] = (bf16)0.0f;
}

// WT[NP][KP] = transpose(W[K][N]) in bf16, zero padded.
__global__ void wt_transpose_kernel(const float* __restrict__ W, bf16* __restrict__ WT,
                                    int K, int N, int KP, int NP) {
  int i = blockIdx.x * blockDim.x + threadIdx.x;
  if (i >= NP * KP) return;
  int nn = i / KP, kk = i - nn * KP;
  float v = (nn < N && kk < K) ? W[(size_t)kk * N + nn] : 0.0f;
  WT[i] = (bf16)v;
}

// ---------------------------------------------------------------------------
// Generic WMMA GEMM: C[M][N] = A[M][K](bf16) * BT[N][K]^T(bf16) + bias (+addend)
// Block = 4 waves, each wave owns one 16-col tile; blockIdx.x = 16-row tile.
// ---------------------------------------------------------------------------
__global__ __launch_bounds__(128) void gemm_bf16_kernel(
    const bf16* __restrict__ A, const bf16* __restrict__ BT,
    const float* __restrict__ bias, const bf16* __restrict__ addend,
    float* __restrict__ Cf, bf16* __restrict__ Cb,
    int N, int K, int lda, int ldbt, int ldc) {
  int wave = threadIdx.x >> 5;
  int lane = threadIdx.x & 31;
  int m0 = blockIdx.x * 16;
  int n0 = (blockIdx.y * 4 + wave) * 16;
  if (n0 >= N) return;  // wave-uniform exit
  const bf16* ab = A + (size_t)m0 * lda;
  const bf16* bb = BT + (size_t)n0 * ldbt;
  v8f c = {};
  for (int k0 = 0; k0 < K; k0 += 32) {
    v16bf a = ld_frag_bf(ab, lda, k0);
    v16bf b = ld_frag_bf(bb, ldbt, k0);
    c = wmma_b(a, b, c);
  }
  int col = lane & 15;
  int rb = (lane & 16) ? 8 : 0;
  float bi = bias ? bias[n0 + col] : 0.0f;
#pragma unroll
  for (int r = 0; r < 8; ++r) {
    size_t o = (size_t)(m0 + rb + r) * ldc + n0 + col;
    float v = c[r] + bi;
    if (addend) v += (float)addend[o];
    if (Cf) Cf[o] = v; else Cb[o] = (bf16)v;
  }
}

// ---------------------------------------------------------------------------
// pair_bias[h][n*1024+m] = (pair[n,m,:] @ Wpb)[h] + bpb[h]   (bf16 output)
// Streams the 268MB pair tensor exactly once. M-tile of 16 (n,m) pairs per wave.
// ---------------------------------------------------------------------------
__global__ __launch_bounds__(256) void pair_bias_kernel(
    const float* __restrict__ pair, const bf16* __restrict__ WpbT,
    const float* __restrict__ bpb, bf16* __restrict__ bias_bf) {
  int wave = threadIdx.x >> 5;
  int lane = threadIdx.x & 31;
  size_t nm0 = ((size_t)blockIdx.x * 8 + wave) * 16;
  v16bf b0 = ld_frag_bf(WpbT, 64, 0);
  v16bf b1 = ld_frag_bf(WpbT, 64, 32);
  const float* ab = pair + nm0 * 64;
  v16bf a0 = ld_frag_f32(ab, 64, 0);
  v16bf a1 = ld_frag_f32(ab, 64, 32);
  v8f c = {};
  c = wmma_b(a0, b0, c);
  c = wmma_b(a1, b1, c);
  int h = lane & 15;
  int rb = (lane & 16) ? 8 : 0;
  if (h < N_H) {  // divergence only after WMMAs
    float bb = bpb[h];
    bf16* dst = bias_bf + (size_t)h * ((size_t)N_SEQ * N_SEQ) + nm0 + rb;
#pragma unroll
    for (int r = 0; r < 8; ++r) dst[r] = (bf16)(c[r] + bb);
  }
}

// ---------------------------------------------------------------------------
// Apply rotation+translation to q/k/v points, compute q2/k2 norms, stage
// bf16 operand layouts (qp/kp row-major padded; v and v_pts transposed).
// One thread per (n, h).
// ---------------------------------------------------------------------------
__global__ __launch_bounds__(256) void transform_kernel(
    const float* __restrict__ qp, const float* __restrict__ kp, const float* __restrict__ vp,
    const float* __restrict__ v, const float* __restrict__ rot, const float* __restrict__ trans,
    bf16* __restrict__ qp_bf, bf16* __restrict__ kp_bf,
    bf16* __restrict__ vpT_bf, bf16* __restrict__ vT_bf,
    float* __restrict__ q2, float* __restrict__ k2) {
  int idx = blockIdx.x * blockDim.x + threadIdx.x;
  if (idx >= N_SEQ * N_H) return;
  int n = idx / N_H, h = idx - n * N_H;
  float R[9], T[3];
#pragma unroll
  for (int i = 0; i < 9; ++i) R[i] = rot[n * 9 + i];
#pragma unroll
  for (int i = 0; i < 3; ++i) T[i] = trans[n * 3 + i];
  float q2s = 0.f, k2s = 0.f;
  bf16* qrow = qp_bf + ((size_t)h * N_SEQ + n) * PD_PAD;
  bf16* krow = kp_bf + ((size_t)h * N_SEQ + n) * PD_PAD;
#pragma unroll
  for (int p = 0; p < 4; ++p) {
    int base = n * 144 + h * 12 + p * 3;
    float qx = qp[base], qy = qp[base + 1], qz = qp[base + 2];
    float kx = kp[base], ky = kp[base + 1], kz = kp[base + 2];
    float vx = vp[base], vy = vp[base + 1], vz = vp[base + 2];
#pragma unroll
    for (int e = 0; e < 3; ++e) {
      float qg = qx * R[e] + qy * R[3 + e] + qz * R[6 + e] + T[e];
      float kg = kx * R[e] + ky * R[3 + e] + kz * R[6 + e] + T[e];
      float vg = vx * R[e] + vy * R[3 + e] + vz * R[6 + e] + T[e];
      qrow[p * 3 + e] = (bf16)qg;
      krow[p * 3 + e] = (bf16)kg;
      vpT_bf[((size_t)h * 16 + p * 3 + e) * N_SEQ + n] = (bf16)vg;
      q2s += qg * qg;
      k2s += kg * kg;
    }
  }
#pragma unroll
  for (int c = P_D; c < PD_PAD; ++c) { qrow[c] = (bf16)0.f; krow[c] = (bf16)0.f; }
#pragma unroll
  for (int r = P_D; r < 16; ++r) vpT_bf[((size_t)h * 16 + r) * N_SEQ + n] = (bf16)0.f;
  q2[h * N_SEQ + n] = q2s;
  k2[h * N_SEQ + n] = k2s;
  const float* vrow = v + (size_t)n * D_S + h * C_H;
#pragma unroll 8
  for (int c = 0; c < C_H; ++c)
    vT_bf[((size_t)h * C_H + c) * N_SEQ + n] = (bf16)vrow[c];
}

// ---------------------------------------------------------------------------
// Fused IPA attention. Block = (head h, 16 query rows), 4 waves.
// Per 256-key chunk: logits via chained WMMA (scalar K=64 + points K=32 into
// one accumulator), finalize with pair bias and -0.5*scale*(q2+k2), chunk-local
// softmax in LDS, then attn@V / attn@V_pts WMMAs accumulated across chunks.
// ---------------------------------------------------------------------------
__global__ __launch_bounds__(128) void ipa_attn_kernel(
    const bf16* __restrict__ q_bf, const bf16* __restrict__ k_bf,
    const bf16* __restrict__ qp_bf, const bf16* __restrict__ kp_bf,
    const bf16* __restrict__ vT_bf, const bf16* __restrict__ vpT_bf,
    const float* __restrict__ q2, const float* __restrict__ k2,
    const bf16* __restrict__ bias_bf,
    bf16* __restrict__ ws_bf, bf16* __restrict__ wp_bf) {
  __shared__ float lg[16 * LSTR];
  __shared__ float red[16][8];
  __shared__ float rowmax[16];
  __shared__ float rowinv[16];
  int h = blockIdx.y;
  int n0 = blockIdx.x * 16;
  int tid = threadIdx.x;
  int wave = tid >> 5, lane = tid & 31;
  const float scale = 0.125f;  // 64^-0.5
  const bf16* qb = q_bf + (size_t)n0 * D_S + h * C_H;
  v16bf aq0 = ld_frag_bf(qb, D_S, 0);
  v16bf aq1 = ld_frag_bf(qb, D_S, 32);
  v16bf aqp = ld_frag_bf(qp_bf + ((size_t)h * N_SEQ + n0) * PD_PAD, PD_PAD, 0);
  v8f accs = {};
  v8f accp = {};
  int col = lane & 15;
  int rb = (lane & 16) ? 8 : 0;
  int srow = tid >> 3, sseg = tid & 7;
  float qterm = -0.5f * scale * q2[h * N_SEQ + n0 + srow];

  for (int ch = 0; ch < NCHUNK; ++ch) {
    int ms = ch * CHK;
    __syncthreads();
    // ---- logits: scale*(q.k + qpts.kpts) via 3 chained WMMAs per tile ----
#pragma unroll
    for (int t = 0; t < 4; ++t) {
      int mt = wave * 4 + t;
      int m0 = ms + mt * 16;
      const bf16* kb = k_bf + (size_t)m0 * D_S + h * C_H;
      v16bf b0 = ld_frag_bf(kb, D_S, 0);
      v16bf b1 = ld_frag_bf(kb, D_S, 32);
      v16bf bp = ld_frag_bf(kp_bf + ((size_t)h * N_SEQ + m0) * PD_PAD, PD_PAD, 0);
      v8f c = {};
      c = wmma_b(aq0, b0, c);
      c = wmma_b(aq1, b1, c);
      c = wmma_b(aqp, bp, c);
#pragma unroll
      for (int r = 0; r < 8; ++r)
        lg[(r + rb) * LSTR + mt * 16 + col] = scale * c[r];
    }
    __syncthreads();
    // ---- finalize logits (+pair bias, -0.5*scale*(q2+k2)) + partial max ----
    {
      const bf16* bi = bias_bf + (size_t)h * N_SEQ * N_SEQ +
                       (size_t)(n0 + srow) * N_SEQ + ms + sseg * 32;
      const float* k2p = k2 + h * N_SEQ + ms + sseg * 32;
      float mx = -3.0e38f;
#pragma unroll 8
      for (int j = 0; j < 32; ++j) {
        int li = srow * LSTR + sseg * 32 + j;
        float val = lg[li] + (float)bi[j] - 0.5f * scale * k2p[j] + qterm;
        lg[li] = val;
        mx = fmaxf(mx, val);
      }
      red[srow][sseg] = mx;
    }
    __syncthreads();
    if (tid < 16) {
      float m = red[tid][0];
#pragma unroll
      for (int s = 1; s < 8; ++s) m = fmaxf(m, red[tid][s]);
      rowmax[tid] = m;
    }
    __syncthreads();
    {
      float m = rowmax[srow];
      float s = 0.f;
#pragma unroll 8
      for (int j = 0; j < 32; ++j) {
        int li = srow * LSTR + sseg * 32 + j;
        float e = __expf(lg[li] - m);
        lg[li] = e;
        s += e;
      }
      red[srow][sseg] = s;
    }
    __syncthreads();
    if (tid < 16) {
      float s = 0.f;
#pragma unroll
      for (int sg = 0; sg < 8; ++sg) s += red[tid][sg];
      rowinv[tid] = 1.0f / s;
    }
    __syncthreads();
    // ---- attn @ V (and attn @ V_pts on wave 0), K=256 in 8 WMMA steps ----
    {
      int c0 = wave * 16;
      const bf16* vb = vT_bf + ((size_t)h * C_H + c0) * N_SEQ + ms;
      const bf16* vpb = vpT_bf + ((size_t)h * 16) * N_SEQ + ms;
      int row = lane & 15;
      float inv = rowinv[row];
      int kb2 = (lane & 16) ? 8 : 0;
#pragma unroll
      for (int ks = 0; ks < 8; ++ks) {
        v16bf a;
        const float* pl = lg + row * LSTR + ks * 32 + kb2;
#pragma unroll
        for (int i = 0; i < 8; ++i) {
          a[i] = (bf16)(pl[i] * inv);
          a[8 + i] = (bf16)(pl[16 + i] * inv);
        }
        v16bf b = ld_frag_bf(vb, N_SEQ, ks * 32);
        accs = wmma_b(a, b, accs);
        if (wave == 0) {  // wave-uniform branch
          v16bf b2 = ld_frag_bf(vpb, N_SEQ, ks * 32);
          accp = wmma_b(a, b2, accp);
        }
      }
    }
  }
  // ---- write weighted_scalar (bf16 [n][768]) and weighted_points ([n][160]) ----
#pragma unroll
  for (int r = 0; r < 8; ++r) {
    int n = n0 + rb + r;
    ws_bf[(size_t)n * D_S + h * C_H + wave * 16 + col] = (bf16)accs[r];
  }
  if (wave == 0 && col < P_D) {
#pragma unroll
    for (int r = 0; r < 8; ++r) {
      int n = n0 + rb + r;
      wp_bf[(size_t)n * K_WP + h * P_D + col] = (bf16)accp[r];
    }
  }
}

// ---------------------------------------------------------------------------
// LayerNorm(single + y) over 768, one block per row.
// ---------------------------------------------------------------------------
__global__ __launch_bounds__(256) void ln_kernel(
    const float* __restrict__ single, const float* __restrict__ y,
    const float* __restrict__ g, const float* __restrict__ b,
    float* __restrict__ out) {
  int n = blockIdx.x;
  int tid = threadIdx.x;
  __shared__ float sred[8];
  size_t base = (size_t)n * D_S;
  float x0 = single[base + tid] + y[base + tid];
  float x1 = single[base + tid + 256] + y[base + tid + 256];
  float x2 = single[base + tid + 512] + y[base + tid + 512];
  float s = x0 + x1 + x2;
#pragma unroll
  for (int o = 16; o > 0; o >>= 1) s += __shfl_down(s, o, 32);
  if ((tid & 31) == 0) sred[tid >> 5] = s;
  __syncthreads();
  float tot = 0.f;
#pragma unroll
  for (int i = 0; i < 8; ++i) tot += sred[i];
  float mu = tot * (1.0f / D_S);
  float d0 = x0 - mu, d1 = x1 - mu, d2 = x2 - mu;
  float vs = d0 * d0 + d1 * d1 + d2 * d2;
#pragma unroll
  for (int o = 16; o > 0; o >>= 1) vs += __shfl_down(vs, o, 32);
  __syncthreads();
  if ((tid & 31) == 0) sred[tid >> 5] = vs;
  __syncthreads();
  float vtot = 0.f;
#pragma unroll
  for (int i = 0; i < 8; ++i) vtot += sred[i];
  float inv = rsqrtf(vtot * (1.0f / D_S) + 1e-5f);
  out[base + tid]       = d0 * inv * g[tid]       + b[tid];
  out[base + tid + 256] = d1 * inv * g[tid + 256] + b[tid + 256];
  out[base + tid + 512] = d2 * inv * g[tid + 512] + b[tid + 512];
}

// ---------------------------------------------------------------------------
extern "C" void kernel_launch(void* const* d_in, const int* in_sizes, int n_in,
                              void* d_out, int out_size, void* d_ws, size_t ws_size,
                              hipStream_t stream) {
  (void)in_sizes; (void)n_in; (void)out_size; (void)ws_size;
  const float* single = (const float*)d_in[0];
  const float* pair   = (const float*)d_in[1];
  const float* rot    = (const float*)d_in[2];
  const float* trans  = (const float*)d_in[3];
  const float* Wq  = (const float*)d_in[4];  const float* bq  = (const float*)d_in[5];
  const float* Wk  = (const float*)d_in[6];  const float* bk  = (const float*)d_in[7];
  const float* Wv  = (const float*)d_in[8];  const float* bv  = (const float*)d_in[9];
  const float* Wpb = (const float*)d_in[10]; const float* bpb = (const float*)d_in[11];
  const float* Wqp = (const float*)d_in[12]; const float* bqp = (const float*)d_in[13];
  const float* Wkp = (const float*)d_in[14]; const float* bkp = (const float*)d_in[15];
  const float* Wvp = (const float*)d_in[16]; const float* bvp = (const float*)d_in[17];
  const float* Wo  = (const float*)d_in[18]; const float* bo  = (const float*)d_in[19];
  const float* Wpo = (const float*)d_in[20]; const float* bpo = (const float*)d_in[21];
  const float* lng = (const float*)d_in[22]; const float* lnb = (const float*)d_in[23];
  float* out = (float*)d_out;

  char* w = (char*)d_ws;
  size_t off = 0;
  auto alloc = [&](size_t bytes) -> void* {
    void* p = w + off;
    off = (off + bytes + 255) & ~(size_t)255;
    return p;
  };
  bf16* single_bf = (bf16*)alloc((size_t)N_SEQ * D_S * 2);
  bf16* WqT  = (bf16*)alloc((size_t)768 * 768 * 2);
  bf16* WkT  = (bf16*)alloc((size_t)768 * 768 * 2);
  bf16* WvT  = (bf16*)alloc((size_t)768 * 768 * 2);
  bf16* WqpT = (bf16*)alloc((size_t)144 * 768 * 2);
  bf16* WkpT = (bf16*)alloc((size_t)144 * 768 * 2);
  bf16* WvpT = (bf16*)alloc((size_t)144 * 768 * 2);
  bf16* WpbT = (bf16*)alloc((size_t)16 * 64 * 2);
  bf16* WpoT = (bf16*)alloc((size_t)768 * K_WP * 2);
  bf16* WoT  = (bf16*)alloc((size_t)768 * 768 * 2);
  bf16* q_bf = (bf16*)alloc((size_t)N_SEQ * D_S * 2);
  bf16* k_bf = (bf16*)alloc((size_t)N_SEQ * D_S * 2);
  float* v_f  = (float*)alloc((size_t)N_SEQ * D_S * 4);
  float* qp_f = (float*)alloc((size_t)N_SEQ * 144 * 4);
  float* kp_f = (float*)alloc((size_t)N_SEQ * 144 * 4);
  float* vp_f = (float*)alloc((size_t)N_SEQ * 144 * 4);
  bf16* qp_bf  = (bf16*)alloc((size_t)N_H * N_SEQ * PD_PAD * 2);
  bf16* kp_bf  = (bf16*)alloc((size_t)N_H * N_SEQ * PD_PAD * 2);
  bf16* vT_bf  = (bf16*)alloc((size_t)N_H * C_H * N_SEQ * 2);
  bf16* vpT_bf = (bf16*)alloc((size_t)N_H * 16 * N_SEQ * 2);
  float* q2 = (float*)alloc((size_t)N_H * N_SEQ * 4);
  float* k2 = (float*)alloc((size_t)N_H * N_SEQ * 4);
  bf16* bias_bf = (bf16*)alloc((size_t)N_H * N_SEQ * N_SEQ * 2);
  bf16* ws_bf  = (bf16*)alloc((size_t)N_SEQ * D_S * 2);
  bf16* wp_bf  = (bf16*)alloc((size_t)N_SEQ * K_WP * 2);
  bf16* tmp_bf = (bf16*)alloc((size_t)N_SEQ * D_S * 2);
  float* y_f = (float*)alloc((size_t)N_SEQ * D_S * 4);

  // 1) convert single, transpose/pad all weights to bf16
  cvt_bf_kernel<<<(N_SEQ * D_S + 255) / 256, 256, 0, stream>>>(single, single_bf, N_SEQ * D_S);
  auto wt = [&](const float* W, bf16* WT, int K, int Nn, int KP, int NPp) {
    int tot = NPp * KP;
    wt_transpose_kernel<<<(tot + 255) / 256, 256, 0, stream>>>(W, WT, K, Nn, KP, NPp);
  };
  wt(Wq, WqT, 768, 768, 768, 768);
  wt(Wk, WkT, 768, 768, 768, 768);
  wt(Wv, WvT, 768, 768, 768, 768);
  wt(Wqp, WqpT, 768, 144, 768, 144);
  wt(Wkp, WkpT, 768, 144, 768, 144);
  wt(Wvp, WvpT, 768, 144, 768, 144);
  wt(Wpb, WpbT, 64, 12, 64, 16);
  wt(Wpo, WpoT, 144, 768, K_WP, 768);
  wt(Wo, WoT, 768, 768, 768, 768);

  // 2) projection GEMMs (WMMA)
  auto gemm = [&](const bf16* A, const bf16* BT, const float* bias, const bf16* addend,
                  float* Cf, bf16* Cb, int M, int Nn, int K, int lda, int ldbt, int ldc) {
    dim3 grid(M / 16, (((Nn + 15) / 16) + 3) / 4);
    gemm_bf16_kernel<<<grid, 128, 0, stream>>>(A, BT, bias, addend, Cf, Cb, Nn, K, lda, ldbt, ldc);
  };
  gemm(single_bf, WqT,  bq,  nullptr, nullptr, q_bf, N_SEQ, 768, 768, 768, 768, 768);
  gemm(single_bf, WkT,  bk,  nullptr, nullptr, k_bf, N_SEQ, 768, 768, 768, 768, 768);
  gemm(single_bf, WvT,  bv,  nullptr, v_f,  nullptr, N_SEQ, 768, 768, 768, 768, 768);
  gemm(single_bf, WqpT, bqp, nullptr, qp_f, nullptr, N_SEQ, 144, 768, 768, 768, 144);
  gemm(single_bf, WkpT, bkp, nullptr, kp_f, nullptr, N_SEQ, 144, 768, 768, 768, 144);
  gemm(single_bf, WvpT, bvp, nullptr, vp_f, nullptr, N_SEQ, 144, 768, 768, 768, 144);

  // 3) rotations + norms + bf16 staging
  transform_kernel<<<(N_SEQ * N_H + 255) / 256, 256, 0, stream>>>(
      qp_f, kp_f, vp_f, v_f, rot, trans, qp_bf, kp_bf, vpT_bf, vT_bf, q2, k2);
  zero_bf_kernel<<<(N_SEQ * K_WP + 255) / 256, 256, 0, stream>>>(wp_bf, N_SEQ * K_WP);

  // 4) pair bias (streams pair once)
  pair_bias_kernel<<<(N_SEQ * N_SEQ / 16) / 8, 256, 0, stream>>>(pair, WpbT, bpb, bias_bf);

  // 5) fused attention
  ipa_attn_kernel<<<dim3(N_SEQ / 16, N_H), 128, 0, stream>>>(
      q_bf, k_bf, qp_bf, kp_bf, vT_bf, vpT_bf, q2, k2, bias_bf, ws_bf, wp_bf);

  // 6) output projections: tmp = ws + wp@Wpo + bpo ; y = tmp@Wo + bo
  gemm(wp_bf, WpoT, bpo, ws_bf, nullptr, tmp_bf, N_SEQ, 768, K_WP, K_WP, K_WP, 768);
  gemm(tmp_bf, WoT, bo, nullptr, y_f, nullptr, N_SEQ, 768, 768, 768, 768, 768);

  // 7) residual layernorm
  ln_kernel<<<N_SEQ, 256, 0, stream>>>(single, y_f, lng, lnb, out);
}
